// LSTMModel_79001628442989
// MI455X (gfx1250) — matrix operand order repaired and checked by reading, not compiled, not run on hardware
//
#include <hip/hip_runtime.h>
#include <hip/hip_bf16.h>

// ---------------------------------------------------------------------------
// Model dims
// ---------------------------------------------------------------------------
#define BB    256          // batch
#define TT    512          // seq len
#define EMBD  300
#define HID   150
#define G4    600          // 4*HID
#define GP    608          // gates padded to 16
#define KP0   320          // EMB padded to 32
#define KP    160          // HID padded to 32
#define MM    (BB*TT)      // 131072 rows
#define NT    38           // GP/16 N-tiles
#define NG    (NT/2)       // 19 N-tile pairs (GEMM)

typedef __attribute__((ext_vector_type(16))) __bf16 v16bf;
typedef __attribute__((ext_vector_type(8)))  __bf16 v8bf;
typedef __attribute__((ext_vector_type(8)))  float  v8f;

static __device__ __forceinline__ v16bf make_a(const __bf16* row, int kb, int kh) {
    // A operand, contiguous k order used here (B matches): lane<16 -> K = {kb..kb+15}
    //                                    lane>=16 -> +16
    v8bf lo = *(const v8bf*)(row + kb + kh * 16);
    v8bf hi = *(const v8bf*)(row + kb + 8 + kh * 16);
    return __builtin_shufflevector(lo, hi, 0,1,2,3,4,5,6,7,8,9,10,11,12,13,14,15);
}

static __device__ __forceinline__ float sigf(float x) {
    return 1.0f / (1.0f + __expf(-x));
}

// Async copy global->LDS (ASYNCcnt path), GVS addressing: mem = SADDR + VADDR.
static __device__ __forceinline__ void async_b128(unsigned ldsOff, int gOff,
                                                  const void* base) {
    asm volatile("global_load_async_to_lds_b128 %0, %1, %2"
                 :: "v"(ldsOff), "v"(gOff), "s"(base) : "memory");
}
static __device__ __forceinline__ void wait_async0() {
    asm volatile("s_wait_asynccnt 0x0" ::: "memory");
}

// ---------------------------------------------------------------------------
// Weight packing: dst[GP][Kp] bf16 = src[N<=G4][K] f32, zero padded
// ---------------------------------------------------------------------------
__global__ void pack_w_kernel(__bf16* __restrict__ dst, const float* __restrict__ src,
                              int N, int K, int Kp) {
    int total = GP * Kp;
    for (int idx = blockIdx.x * blockDim.x + threadIdx.x; idx < total;
         idx += gridDim.x * blockDim.x) {
        int n = idx / Kp, k = idx % Kp;
        float v = (n < N && k < K) ? src[(size_t)n * K + k] : 0.0f;
        dst[idx] = (__bf16)v;
    }
}

__global__ void pack_bias_kernel(float* __restrict__ dst,
                                 const float* __restrict__ bih,
                                 const float* __restrict__ bhh) {
    int n = blockIdx.x * blockDim.x + threadIdx.x;
    if (n < GP) dst[n] = (n < G4) ? (bih[n] + bhh[n]) : 0.0f;
}

// ---------------------------------------------------------------------------
// Embedding gather -> bf16 [MM][KP0], zero padded cols
// ---------------------------------------------------------------------------
__global__ void embed_kernel(__bf16* __restrict__ dst,
                             const float* __restrict__ emb,
                             const int* __restrict__ x) {
    size_t total = (size_t)MM * KP0;
    for (size_t idx = (size_t)blockIdx.x * blockDim.x + threadIdx.x; idx < total;
         idx += (size_t)gridDim.x * blockDim.x) {
        size_t m = idx / KP0;
        int    k = (int)(idx % KP0);
        int tok = x[m];
        float v = (k < EMBD) ? emb[(size_t)tok * EMBD + k] : 0.0f;
        dst[idx] = (__bf16)v;
    }
}

// ---------------------------------------------------------------------------
// Input projection GEMM: xg[MM][GP] = A[MM][lda](bf16) @ Bw[GP][lda]^T + bias
// 8 waves/block; each wave: one 16-row A tile x TWO 16-col N-tiles (A reuse).
// grid.x = (MM/16/8) * NG
// ---------------------------------------------------------------------------
__global__ void __launch_bounds__(256) xproj_gemm_kernel(
    const __bf16* __restrict__ A, int lda,
    const __bf16* __restrict__ Bw,
    const float*  __restrict__ bias,
    float* __restrict__ xg, int ksteps) {
    int wave = threadIdx.x >> 5;
    int lane = threadIdx.x & 31;
    int ng    = blockIdx.x % NG;
    int mtile = (blockIdx.x / NG) * 8 + wave;
    int nt0 = ng * 2, nt1 = ng * 2 + 1;
    int mr = lane & 15;       // A row / B col / D col
    int kh = lane >> 4;       // K-half selector

    float b0 = bias[nt0 * 16 + mr];
    float b1 = bias[nt1 * 16 + mr];
    v8f acc0, acc1;
#pragma unroll
    for (int v = 0; v < 8; ++v) { acc0[v] = b0; acc1[v] = b1; }

    const __bf16* arow  = A  + (size_t)(mtile * 16 + mr) * lda;
    const __bf16* brow0 = Bw + (size_t)(nt0 * 16 + mr) * lda + kh * 16;
    const __bf16* brow1 = Bw + (size_t)(nt1 * 16 + mr) * lda + kh * 16;

    for (int ks = 0; ks < ksteps; ++ks) {
        int kb = ks * 32;
        v16bf a  = make_a(arow, kb, kh);
        v16bf bv0 = *(const v16bf*)(brow0 + kb);
        v16bf bv1 = *(const v16bf*)(brow1 + kb);
        acc0 = __builtin_amdgcn_wmma_f32_16x16x32_bf16(false, a, false, bv0,
                                                       (short)0, acc0, false, false);
        acc1 = __builtin_amdgcn_wmma_f32_16x16x32_bf16(false, a, false, bv1,
                                                       (short)0, acc1, false, false);
    }
    // D: element v -> row mtile*16 + v + 8*kh, col ntile*16 + mr
    float* out0 = xg + (size_t)(mtile * 16 + 8 * kh) * GP + nt0 * 16 + mr;
    float* out1 = xg + (size_t)(mtile * 16 + 8 * kh) * GP + nt1 * 16 + mr;
#pragma unroll
    for (int v = 0; v < 8; ++v) { out0[(size_t)v * GP] = acc0[v];
                                  out1[(size_t)v * GP] = acc1[v]; }
}

// ---------------------------------------------------------------------------
// Recurrent kernel: one block = 16 batch rows, full time loop.
// LDS: Whh [GP][KP] bf16 | h bf16 [16][KP] | xg stage x2 [16][GP] f32
//      | c f32 [16][152] | lengths [16]
// Next step's xg slab is async-prefetched into the other stage buffer while
// WMMAs for the current step run; stage buffer doubles as the gates buffer.
// ---------------------------------------------------------------------------
#define CSTR 152
#define STG  (16*GP)     // floats per stage buffer
#define REC_LDS ((size_t)GP*KP*2 + 16*KP*2 + (size_t)2*STG*4 + 16*CSTR*4 + 16*4)

__global__ void __launch_bounds__(256) lstm_rec_kernel(
    const float*  __restrict__ xg,      // [MM][GP], m = b*TT + t
    const __bf16* __restrict__ Whh,     // [GP][KP] packed bf16
    const int*    __restrict__ lengths, // [BB]
    __bf16* __restrict__ yout,          // [MM][KP] or null
    float*  __restrict__ hn) {          // [BB][HID] or null
    extern __shared__ char smem[];
    __bf16* wB   = (__bf16*)smem;
    __bf16* hbf  = (__bf16*)(smem + (size_t)GP * KP * 2);
    float*  xstg = (float*) (smem + (size_t)GP * KP * 2 + 16 * KP * 2);
    float*  cst  = xstg + 2 * STG;
    int*    lsm  = (int*)(cst + 16 * CSTR);

    int tid  = threadIdx.x;
    int wave = tid >> 5;
    int lane = tid & 31;
    int row0 = blockIdx.x * 16;      // first batch row of this block
    int mr = lane & 15;
    int kh = lane >> 4;

    // Async slab fetch: 16 rows x GP floats (2432 B/row = 152 b128 chunks)
    auto load_slab = [&](int t, float* buf) {
        unsigned lbase = (unsigned)(size_t)buf;
        for (int i = tid; i < 16 * 152; i += 256) {
            int r = i / 152, c = i % 152;
            unsigned loff = lbase + (unsigned)(r * GP * 4 + c * 16);
            int goff = (int)((((size_t)(row0 + r) * TT + t) * GP) * 4 + (size_t)c * 16);
            async_b128(loff, goff, xg);
        }
    };

    // one-time: Whh -> LDS (async), slab 0 -> stage 0, zero h/c, stage lengths
    {
        unsigned lbase = (unsigned)(size_t)wB;
        int n16 = GP * KP * 2 / 16;              // 12160 chunks
        for (int i = tid; i < n16; i += 256)
            async_b128(lbase + (unsigned)(i * 16), i * 16, Whh);
    }
    load_slab(0, xstg);
    for (int i = tid; i < 16 * KP; i += 256) hbf[i] = (__bf16)0.0f;
    for (int i = tid; i < 16 * CSTR; i += 256) cst[i] = 0.0f;
    if (tid < 16) lsm[tid] = lengths[row0 + tid];
    wait_async0();
    __syncthreads();

    for (int t = 0; t < TT; ++t) {
        float* xcur = xstg + (t & 1) * STG;
        // prefetch next step's slab while this step computes
        if (t + 1 < TT) load_slab(t + 1, xstg + ((t + 1) & 1) * STG);

        // ---- phase A: gates = xg[:,t,:] + h @ Whh^T (WMMA, in-place on xcur)
        for (int nt = wave; nt < NT; nt += 8) {
            float* xp = xcur + (8 * kh) * GP + nt * 16 + mr;
            v8f acc;
#pragma unroll
            for (int v = 0; v < 8; ++v) acc[v] = xp[v * GP];
            const __bf16* hrow = hbf + mr * KP;
            const __bf16* brow = wB + (nt * 16 + mr) * KP + kh * 16;
#pragma unroll
            for (int ks = 0; ks < KP / 32; ++ks) {
                int kb = ks * 32;
                v16bf a = make_a(hrow, kb, kh);
                v16bf b = *(const v16bf*)(brow + kb);
                acc = __builtin_amdgcn_wmma_f32_16x16x32_bf16(false, a, false, b,
                                                              (short)0, acc, false, false);
            }
#pragma unroll
            for (int v = 0; v < 8; ++v) xp[v * GP] = acc[v];
        }
        __syncthreads();

        // ---- phase B: elementwise LSTM cell + masked state update ----
        for (int idx = tid; idx < 16 * KP; idx += 256) {
            int r = idx / KP, u = idx % KP;
            size_t ym = ((size_t)(row0 + r) * TT + t) * KP + u;
            if (u < HID) {
                float gi = xcur[r * GP + u];
                float gf = xcur[r * GP + HID + u];
                float gg = xcur[r * GP + 2 * HID + u];
                float go = xcur[r * GP + 3 * HID + u];
                float i_ = sigf(gi), f_ = sigf(gf), o_ = sigf(go);
                float g_ = tanhf(gg);
                float cold = cst[r * CSTR + u];
                float cnew = f_ * cold + i_ * g_;
                float hnew = o_ * tanhf(cnew);
                bool mask = t < lsm[r];
                cst[r * CSTR + u] = mask ? cnew : cold;
                float hold = (float)hbf[r * KP + u];
                hbf[r * KP + u] = (__bf16)(mask ? hnew : hold);
                if (yout) yout[ym] = (__bf16)(mask ? hnew : 0.0f);
            } else {
                if (yout) yout[ym] = (__bf16)0.0f;   // keep K-padding zero
            }
        }
        wait_async0();     // next slab resident before anyone reads it
        __syncthreads();
    }

    if (hn) {
        for (int idx = tid; idx < 16 * HID; idx += 256) {
            int r = idx / HID, u = idx % HID;
            hn[(size_t)(row0 + r) * HID + u] = (float)hbf[r * KP + u];
        }
    }
}

// ---------------------------------------------------------------------------
// FC head: out[b] = fc2_w . relu(fc1_w @ hn[b] + fc1_b) + fc2_b
// ---------------------------------------------------------------------------
__global__ void fc_kernel(const float* __restrict__ hn,
                          const float* __restrict__ w1, const float* __restrict__ b1,
                          const float* __restrict__ w2, const float* __restrict__ b2,
                          float* __restrict__ out) {
    int b = blockIdx.x * blockDim.x + threadIdx.x;
    if (b >= BB) return;
    const float* h = hn + (size_t)b * HID;
    float o = b2[0];
    for (int j = 0; j < HID; ++j) {
        float z = b1[j];
        const float* wr = w1 + (size_t)j * HID;
        for (int k = 0; k < HID; ++k) z = fmaf(h[k], wr[k], z);
        z = z > 0.0f ? z : 0.0f;
        o = fmaf(z, w2[j], o);
    }
    out[b] = o;
}

// ---------------------------------------------------------------------------
// Host side
// ---------------------------------------------------------------------------
static size_t align256(size_t x) { return (x + 255) & ~(size_t)255; }

extern "C" void kernel_launch(void* const* d_in, const int* in_sizes, int n_in,
                              void* d_out, int out_size, void* d_ws, size_t ws_size,
                              hipStream_t stream) {
    (void)in_sizes; (void)n_in; (void)out_size; (void)ws_size;
    const int*   x       = (const int*)  d_in[0];
    const int*   lengths = (const int*)  d_in[1];
    const float* emb     = (const float*)d_in[2];
    const float* Wih[4], *Whh[4], *bih[4], *bhh[4];
    for (int l = 0; l < 4; ++l) {
        Wih[l] = (const float*)d_in[3 + 4 * l];
        Whh[l] = (const float*)d_in[4 + 4 * l];
        bih[l] = (const float*)d_in[5 + 4 * l];
        bhh[l] = (const float*)d_in[6 + 4 * l];
    }
    const float* fc1_w = (const float*)d_in[19];
    const float* fc1_b = (const float*)d_in[20];
    const float* fc2_w = (const float*)d_in[21];
    const float* fc2_b = (const float*)d_in[22];
    float* out = (float*)d_out;

    // workspace carve-up
    char* ws = (char*)d_ws;
    size_t off = 0;
    auto carve = [&](size_t bytes) { char* p = ws + off; off += align256(bytes); return p; };
    __bf16* y0 = (__bf16*)carve((size_t)MM * KP0 * 2);   // emb bf16
    __bf16* y1 = (__bf16*)carve((size_t)MM * KP  * 2);
    __bf16* y2 = (__bf16*)carve((size_t)MM * KP  * 2);
    float*  xg = (float*) carve((size_t)MM * GP  * 4);
    __bf16* pWih[4]; __bf16* pWhh[4]; float* pBias[4];
    int Kin[4] = {KP0, KP, KP, KP};
    for (int l = 0; l < 4; ++l) {
        pWih[l]  = (__bf16*)carve((size_t)GP * Kin[l] * 2);
        pWhh[l]  = (__bf16*)carve((size_t)GP * KP * 2);
        pBias[l] = (float*) carve((size_t)GP * 4);
    }
    float* hn = (float*)carve((size_t)BB * HID * 4);

    // allow big dynamic LDS for the recurrent kernel
    (void)hipFuncSetAttribute((const void*)lstm_rec_kernel,
                              hipFuncAttributeMaxDynamicSharedMemorySize,
                              (int)REC_LDS);

    // pack weights / biases
    int Ksrc[4] = {EMBD, HID, HID, HID};
    for (int l = 0; l < 4; ++l) {
        pack_w_kernel<<<256, 256, 0, stream>>>(pWih[l], Wih[l], G4, Ksrc[l], Kin[l]);
        pack_w_kernel<<<256, 256, 0, stream>>>(pWhh[l], Whh[l], G4, HID, KP);
        pack_bias_kernel<<<(GP + 255) / 256, 256, 0, stream>>>(pBias[l], bih[l], bhh[l]);
    }

    // embedding gather -> bf16
    embed_kernel<<<4096, 256, 0, stream>>>(y0, emb, x);

    const int gemm_grid = (MM / 16 / 8) * NG;   // 1024 * 19

    // layer 0
    xproj_gemm_kernel<<<gemm_grid, 256, 0, stream>>>(y0, KP0, pWih[0], pBias[0], xg, KP0 / 32);
    lstm_rec_kernel<<<BB / 16, 256, REC_LDS, stream>>>(xg, pWhh[0], lengths, y1, nullptr);
    // layer 1
    xproj_gemm_kernel<<<gemm_grid, 256, 0, stream>>>(y1, KP, pWih[1], pBias[1], xg, KP / 32);
    lstm_rec_kernel<<<BB / 16, 256, REC_LDS, stream>>>(xg, pWhh[1], lengths, y2, nullptr);
    // layer 2
    xproj_gemm_kernel<<<gemm_grid, 256, 0, stream>>>(y2, KP, pWih[2], pBias[2], xg, KP / 32);
    lstm_rec_kernel<<<BB / 16, 256, REC_LDS, stream>>>(xg, pWhh[2], lengths, y1, nullptr);
    // layer 3 (only hn needed)
    xproj_gemm_kernel<<<gemm_grid, 256, 0, stream>>>(y1, KP, pWih[3], pBias[3], xg, KP / 32);
    lstm_rec_kernel<<<BB / 16, 256, REC_LDS, stream>>>(xg, pWhh[3], lengths, nullptr, hn);

    // FC head
    fc_kernel<<<1, 256, 0, stream>>>(hn, fc1_w, fc1_b, fc2_w, fc2_b, out);
}
